// _SwinV2Attention_9809705304232
// MI455X (gfx1250) — matrix operand order repaired and checked by reading, not compiled, not run on hardware
//
#include <hip/hip_runtime.h>
#include <hip/hip_bf16.h>
#include <math.h>

typedef __attribute__((ext_vector_type(16))) _Float16 v16h;
typedef __attribute__((ext_vector_type(8)))  _Float16 v8h;
typedef __attribute__((ext_vector_type(8)))  float    v8f;

#define WS   7
#define NH   12
#define HD   32
#define DIM  384
#define NTOK 49          // tokens per window
#define NN   (NTOK*NTOK) // 2401

// ---------------------------------------------------------------------------
// WMMA helpers (CDNA5 gfx1250, wave32)
// ---------------------------------------------------------------------------
__device__ __forceinline__ v8f wmma_f16(v16h a, v16h b, v8f c) {
    // D = A(16x32 f16) * B(32x16 f16) + C(16x16 f32)
    return __builtin_amdgcn_wmma_f32_16x16x32_f16(
        /*neg_a=*/false, a, /*neg_b=*/false, b,
        /*c_mod=*/(short)0, c, /*reuse_a=*/false, /*reuse_b=*/false);
}

// A fragment: 16x32, row-major source with leading dim ld (elements).
// lane<16  holds row m=lane   : K = [k0..k0+7] , [k0+16..k0+23]
// lane>=16 holds row m=lane-16: K = [k0+8..+15], [k0+24..+31]
__device__ __forceinline__ v16h load_frag_A(const _Float16* base, int row0,
                                            int k0, int ld, int lane) {
    int half = lane >> 4;
    int m    = row0 + (lane & 15);
    const _Float16* p = base + (size_t)m * ld + k0 + half * 8;
    v8h lo = *(const v8h*)(p);
    v8h hi = *(const v8h*)(p + 16);
    v16h r;
#pragma unroll
    for (int i = 0; i < 8; ++i) { r[i] = lo[i]; r[i + 8] = hi[i]; }
    return r;
}

// Same but clamps the row index (for the 49->64 padding on global x).
__device__ __forceinline__ v16h load_frag_A_clamp(const _Float16* base, int row0,
                                                  int k0, int ld, int lane,
                                                  int maxrow) {
    int half = lane >> 4;
    int m    = row0 + (lane & 15);
    if (m > maxrow) m = maxrow;
    const _Float16* p = base + (size_t)m * ld + k0 + half * 8;
    v8h lo = *(const v8h*)(p);
    v8h hi = *(const v8h*)(p + 16);
    v16h r;
#pragma unroll
    for (int i = 0; i < 8; ++i) { r[i] = lo[i]; r[i + 8] = hi[i]; }
    return r;
}

// B fragment: 32x16 (KxN). Source stores each output column as a row of
// contiguous K values with leading dim ld ("weights row-major": row=feature).
// lane<16 holds col=lane, K=[k0..k0+7],[k0+16..k0+23]; lane>=16 holds col, K=[k0+8..+15],[k0+24..+31]
__device__ __forceinline__ v16h load_frag_B(const _Float16* base, int col0,
                                            int k0, int ld, int lane) {
    int c  = lane & 15;
    int kb = k0 + (lane >> 4) * 8;
    const _Float16* p = base + (size_t)(col0 + c) * ld + kb;
    v8h lo = *(const v8h*)(p);
    v8h hi = *(const v8h*)(p + 16);
    v16h r;
#pragma unroll
    for (int i = 0; i < 8; ++i) { r[i] = lo[i]; r[i + 8] = hi[i]; }
    return r;
}

// ---------------------------------------------------------------------------
// f32 -> f16 conversion (grid-stride)
// ---------------------------------------------------------------------------
__global__ void cvt_f32_to_f16(const float* __restrict__ in,
                               _Float16* __restrict__ out, int n) {
    int i = blockIdx.x * blockDim.x + threadIdx.x;
    int stride = gridDim.x * blockDim.x;
    for (; i < n; i += stride) out[i] = (_Float16)in[i];
}

// ---------------------------------------------------------------------------
// Continuous position bias: cpb[h, i, j] for all (i, j) token pairs
// ---------------------------------------------------------------------------
__global__ void cpb_kernel(const float* __restrict__ w1, const float* __restrict__ b1,
                           const float* __restrict__ w2, const float* __restrict__ b2,
                           float* __restrict__ cpb) {
    int t = blockIdx.x * blockDim.x + threadIdx.x;
    if (t >= NN) return;
    int i = t / NTOK, j = t % NTOK;
    float dy = (float)(i / WS - j / WS);
    float dx = (float)(i % WS - j % WS);
    float r0 = copysignf(log1pf(fabsf(dy)), dy);
    float r1 = copysignf(log1pf(fabsf(dx)), dx);
    float hid[64];
#pragma unroll
    for (int k = 0; k < 64; ++k)
        hid[k] = fmaxf(r0 * w1[2 * k] + r1 * w1[2 * k + 1] + b1[k], 0.0f);
    for (int h = 0; h < NH; ++h) {
        float s = b2[h];
#pragma unroll
        for (int k = 0; k < 64; ++k) s += hid[k] * w2[h * 64 + k];
        cpb[h * NN + t] = s;
    }
}

// ---------------------------------------------------------------------------
// Fused per-(window, head) attention: QKV GEMM + cosine-norm + S + bias +
// softmax + P@V, all via v_wmma_f32_16x16x32_f16.  grid = (B_, NH), 256 thr.
// ---------------------------------------------------------------------------
#define SQ_LD  40   // q/k f16 LDS stride (conflict-free for b128 frag loads)
#define SVT_LD 72   // v-transposed f16 LDS stride
#define SS_LD  68   // f32 score / staging stride
#define SP_LD  72   // probability f16 stride

__global__ __launch_bounds__(256)
void attn_kernel(const _Float16* __restrict__ x16,    // (B_*49, 384)
                 const _Float16* __restrict__ qkvw16, // (1152, 384)
                 const float*    __restrict__ qkv_b,  // 1152
                 const float*    __restrict__ tau,    // NH
                 const float*    __restrict__ cpb,    // (NH, 49, 49)
                 const float*    __restrict__ mask,   // (nW, 49, 49)
                 int nW,
                 _Float16*       __restrict__ y16)    // (B_*49, 384)
{
    __shared__ _Float16 sQ[64 * SQ_LD];
    __shared__ _Float16 sK[64 * SQ_LD];
    __shared__ _Float16 sVT[HD * SVT_LD];
    __shared__ float    sS[64 * SS_LD];
    __shared__ _Float16 sP[64 * SP_LD];

    const int bw   = blockIdx.x;        // window
    const int h    = blockIdx.y;        // head
    const int tid  = threadIdx.x;
    const int wave = tid >> 5;
    const int lane = tid & 31;

    const _Float16* xwin = x16 + (size_t)bw * NTOK * DIM;

    // ---- Stage 1: QKV sub-GEMM for this head.  24 frags = 8 waves * 3.
    // frag f: c = f/8 (0=q,1=k,2=v), mt = (f%8)/2, nt = (f%8)%2
#pragma unroll
    for (int i = 0; i < 3; ++i) {
        int f  = wave * 3 + i;
        int c  = f >> 3;
        int rm = f & 7;
        int mt = rm >> 1, nt = rm & 1;
        int brow0 = c * DIM + h * HD + nt * 16;   // weight rows (features)
        v8f acc;
#pragma unroll
        for (int r = 0; r < 8; ++r) acc[r] = 0.0f;
#pragma unroll
        for (int k0 = 0; k0 < DIM; k0 += 32) {
            v16h a = load_frag_A_clamp(xwin, mt * 16, k0, DIM, lane, NTOK - 1);
            v16h b = load_frag_B(qkvw16, brow0, k0, DIM, lane);
            acc = wmma_f16(a, b, acc);
        }
        int col  = lane & 15;
        int hm   = (lane >> 4) * 8;
        int d    = nt * 16 + col;
        float bi = qkv_b[brow0 + col];
        if (c == 2) {          // V: store transposed f16 (row=d, col=token)
#pragma unroll
            for (int r = 0; r < 8; ++r) {
                int m = mt * 16 + hm + r;
                sVT[d * SVT_LD + m] = (_Float16)(acc[r] + bi);
            }
        } else {               // Q/K: stage f32 for row normalization
            int cc = c * 32 + d;
#pragma unroll
            for (int r = 0; r < 8; ++r) {
                int m = mt * 16 + hm + r;
                sS[m * SS_LD + cc] = acc[r] + bi;
            }
        }
    }
    __syncthreads();

    // ---- Stage 2: cosine normalization of q and k rows -> f16
    if (tid < 128) {
        int row   = tid & 63;
        int which = tid >> 6;                 // 0 = q, 1 = k
        _Float16* dst = which ? sK : sQ;
        if (row < NTOK) {
            float s = 0.0f;
            float v[HD];
#pragma unroll
            for (int d = 0; d < HD; ++d) {
                float x = sS[row * SS_LD + which * 32 + d];
                v[d] = x; s += x * x;
            }
            float inv = 1.0f / fmaxf(sqrtf(s), 1e-12f);
#pragma unroll
            for (int d = 0; d < HD; ++d)
                dst[row * SQ_LD + d] = (_Float16)(v[d] * inv);
        } else {
#pragma unroll
            for (int d = 0; d < HD; ++d)
                dst[row * SQ_LD + d] = (_Float16)0.0f;
        }
    }
    __syncthreads();

    // ---- Stage 3: S = q_hat @ k_hat^T  (64x64, K=32), + tau/cpb/mask
    const float scale = 1.0f / fmaxf(tau[h], 0.01f);
    const int   wi    = bw % nW;
#pragma unroll
    for (int i = 0; i < 2; ++i) {
        int f  = wave * 2 + i;
        int mt = f >> 2, nt = f & 3;
        v8f acc;
#pragma unroll
        for (int r = 0; r < 8; ++r) acc[r] = 0.0f;
        v16h a = load_frag_A(sQ, mt * 16, 0, SQ_LD, lane);
        v16h b = load_frag_B(sK, nt * 16, 0, SQ_LD, lane);
        acc = wmma_f16(a, b, acc);
        int n  = nt * 16 + (lane & 15);
        int hm = (lane >> 4) * 8;
#pragma unroll
        for (int r = 0; r < 8; ++r) {
            int m = mt * 16 + hm + r;
            float val;
            if (m < NTOK && n < NTOK) {
                val = acc[r] * scale + cpb[h * NN + m * NTOK + n]
                                     + mask[wi * NN + m * NTOK + n];
            } else {
                val = -1e30f;    // padded key columns never win softmax
            }
            sS[m * SS_LD + n] = val;
        }
    }
    __syncthreads();

    // ---- Stage 4: row softmax -> P (f16), padded rows/cols zeroed
    if (tid < 64) {
        int m = tid;
        if (m < NTOK) {
            float mx = -1e30f;
            for (int n = 0; n < NTOK; ++n) mx = fmaxf(mx, sS[m * SS_LD + n]);
            float sum = 0.0f;
            for (int n = 0; n < NTOK; ++n) sum += __expf(sS[m * SS_LD + n] - mx);
            float inv = 1.0f / sum;
            for (int n = 0; n < 64; ++n) {
                float p = (n < NTOK) ? __expf(sS[m * SS_LD + n] - mx) * inv : 0.0f;
                sP[m * SP_LD + n] = (_Float16)p;
            }
        } else {
            for (int n = 0; n < 64; ++n) sP[m * SP_LD + n] = (_Float16)0.0f;
        }
    }
    __syncthreads();

    // ---- Stage 5: y = P @ V   (64x32, K=64), 8 frags = 1 per wave
    {
        int mt = wave >> 1, nt = wave & 1;
        v8f acc;
#pragma unroll
        for (int r = 0; r < 8; ++r) acc[r] = 0.0f;
#pragma unroll
        for (int k0 = 0; k0 < 64; k0 += 32) {
            v16h a = load_frag_A(sP, mt * 16, k0, SP_LD, lane);
            v16h b = load_frag_B(sVT, nt * 16, k0, SVT_LD, lane);
            acc = wmma_f16(a, b, acc);
        }
        int d  = nt * 16 + (lane & 15);
        int hm = (lane >> 4) * 8;
#pragma unroll
        for (int r = 0; r < 8; ++r) {
            int m = mt * 16 + hm + r;
            if (m < NTOK)
                y16[((size_t)bw * NTOK + m) * DIM + h * HD + d] = (_Float16)acc[r];
        }
    }
}

// ---------------------------------------------------------------------------
// Output projection: out(M,384) = Y16(M,384) @ proj_w^T + proj_b  (f32 out)
// Each wave: one 16-row tile x 128 columns (8 WMMA accumulators).
// grid = (Mtiles/8, 384/128), 256 threads.
// ---------------------------------------------------------------------------
__global__ __launch_bounds__(256)
void proj_kernel(const _Float16* __restrict__ y16,
                 const _Float16* __restrict__ pw16,   // (384, 384) row=out feat
                 const float*    __restrict__ pb,
                 float*          __restrict__ out)
{
    const int wave = threadIdx.x >> 5;
    const int lane = threadIdx.x & 31;
    const int m0   = (blockIdx.x * 8 + wave) * 16;
    const int c0   = blockIdx.y * 128;

    v8f acc[8];
#pragma unroll
    for (int t = 0; t < 8; ++t)
#pragma unroll
        for (int r = 0; r < 8; ++r) acc[t][r] = 0.0f;

    for (int k0 = 0; k0 < DIM; k0 += 32) {
        v16h a = load_frag_A(y16, m0, k0, DIM, lane);
#pragma unroll
        for (int nt = 0; nt < 8; ++nt) {
            v16h b = load_frag_B(pw16, c0 + nt * 16, k0, DIM, lane);
            acc[nt] = wmma_f16(a, b, acc[nt]);
        }
    }
    int hm = (lane >> 4) * 8;
#pragma unroll
    for (int nt = 0; nt < 8; ++nt) {
        int col  = c0 + nt * 16 + (lane & 15);
        float bi = pb[col];
#pragma unroll
        for (int r = 0; r < 8; ++r)
            out[(size_t)(m0 + hm + r) * DIM + col] = acc[nt][r] + bi;
    }
}

// ---------------------------------------------------------------------------
// Host launcher
// ---------------------------------------------------------------------------
extern "C" void kernel_launch(void* const* d_in, const int* in_sizes, int n_in,
                              void* d_out, int out_size, void* d_ws, size_t ws_size,
                              hipStream_t stream) {
    const float* x      = (const float*)d_in[0];
    const float* mask   = (const float*)d_in[1];
    const float* qkv_w  = (const float*)d_in[2];
    const float* qkv_b  = (const float*)d_in[3];
    const float* tau    = (const float*)d_in[4];
    const float* cpb_w1 = (const float*)d_in[5];
    const float* cpb_b1 = (const float*)d_in[6];
    const float* cpb_w2 = (const float*)d_in[7];
    const float* cpb_b2 = (const float*)d_in[8];
    const float* proj_w = (const float*)d_in[9];
    const float* proj_b = (const float*)d_in[10];
    float* out = (float*)d_out;

    const int B_ = in_sizes[0] / (NTOK * DIM);   // 2048
    const int nW = in_sizes[1] / NN;             // 64
    const int M  = B_ * NTOK;                    // 100352 (multiple of 16)

    // Workspace layout (f16 buffers + cpb table), 256B aligned
    char* w = (char*)d_ws;
    size_t o = 0;
    _Float16* x16  = (_Float16*)(w + o); o += (size_t)M * DIM * 2;       o = (o + 255) & ~(size_t)255;
    _Float16* y16  = (_Float16*)(w + o); o += (size_t)M * DIM * 2;       o = (o + 255) & ~(size_t)255;
    _Float16* qw16 = (_Float16*)(w + o); o += (size_t)3 * DIM * DIM * 2; o = (o + 255) & ~(size_t)255;
    _Float16* pw16 = (_Float16*)(w + o); o += (size_t)DIM * DIM * 2;     o = (o + 255) & ~(size_t)255;
    float*    cpbt = (float*)(w + o);    o += (size_t)NH * NN * 4;
    (void)ws_size; (void)n_in; (void)out_size;

    // 1) f32 -> f16 conversions
    {
        int n = M * DIM;
        int blocks = (n + 255) / 256; if (blocks > 8192) blocks = 8192;
        cvt_f32_to_f16<<<blocks, 256, 0, stream>>>(x, x16, n);
    }
    {
        int n = 3 * DIM * DIM;
        cvt_f32_to_f16<<<(n + 255) / 256, 256, 0, stream>>>(qkv_w, qw16, n);
    }
    {
        int n = DIM * DIM;
        cvt_f32_to_f16<<<(n + 255) / 256, 256, 0, stream>>>(proj_w, pw16, n);
    }

    // 2) CPB bias table
    cpb_kernel<<<(NN + 255) / 256, 256, 0, stream>>>(cpb_w1, cpb_b1, cpb_w2, cpb_b2, cpbt);

    // 3) fused attention (QKV + cosine attn + softmax + P@V)
    attn_kernel<<<dim3(B_, NH), 256, 0, stream>>>(x16, qw16, qkv_b, tau, cpbt,
                                                  mask, nW, y16);

    // 4) output projection
    proj_kernel<<<dim3(M / 16 / 8, DIM / 128), 256, 0, stream>>>(y16, pw16, proj_b, out);
}
